// MoE_87832081203672
// MI455X (gfx1250) — matrix operand order repaired and checked by reading, not compiled, not run on hardware
//
#include <hip/hip_runtime.h>
#include <hip/hip_bf16.h>
#include <math.h>

// Problem constants (match reference)
#define B_TOK 8192
#define DDIM  1024
#define HDIM  4096
#define NEXP  8

// Tiling
#define TM       64                    // tokens per workgroup tile
#define HC       128                   // H-chunk width (hidden staged in LDS)
#define THREADS  512                   // 16 waves (wave32)
#define TILES_PER_E (B_TOK / TM)       // 128 (worst case: expert gets all tokens)

typedef __attribute__((ext_vector_type(16))) __bf16       v16bf;
typedef __attribute__((ext_vector_type(8)))  float        v8f;
typedef __attribute__((ext_vector_type(8)))  unsigned int v8u;

// ---------- helpers ----------
__device__ __forceinline__ unsigned short f2bf(float f) {
  unsigned u = __float_as_uint(f);
  u += 0x7FFFu + ((u >> 16) & 1u);          // round-to-nearest-even
  return (unsigned short)(u >> 16);
}
__device__ __forceinline__ unsigned pack2(float a, float b) {
  return (unsigned)f2bf(a) | ((unsigned)f2bf(b) << 16);
}
__device__ __forceinline__ float gelu_exact(float v) {
  return 0.5f * v * (1.0f + erff(v * 0.70710678118654752440f));
}
__device__ __forceinline__ v8f vzero8() {
  v8f z;
#pragma unroll
  for (int j = 0; j < 8; ++j) z[j] = 0.0f;
  return z;
}
__device__ __forceinline__ v8f wmma_bf16(v16bf a, v16bf b, v8f c) {
  return __builtin_amdgcn_wmma_f32_16x16x32_bf16(false, a, false, b, (short)0, c,
                                                 false, false);
}

// A fragment (16x32 bf16, ISA 7.12.2) from a row-major bf16 row.
// Lane holds M-row = lane&15; lanes>=16 shift K by +8.
//   V0..3  <- bf16[k0 + hi*8 .. +7]   (16 contiguous bytes)
//   V4..7  <- bf16[k0 + 16 + hi*8 .. +7]
// Works for both global (xbf) and LDS (hs) pointers.
__device__ __forceinline__ v16bf load_a16(const unsigned short* row, int k0,
                                          int lane_hi) {
  const char* p = (const char*)row + (size_t)(k0 + lane_hi * 8) * 2;
  uint4 lo = *(const uint4*)p;
  uint4 hi = *(const uint4*)(p + 32);
  v8u u;
  u[0] = lo.x; u[1] = lo.y; u[2] = lo.z; u[3] = lo.w;
  u[4] = hi.x; u[5] = hi.y; u[6] = hi.z; u[7] = hi.w;
  return __builtin_bit_cast(v16bf, u);
}

// B fragment from pre-swizzled weights: one 1KB block per 32x16 (KxN) tile,
// stored as [lane][8 dwords] exactly matching the B VGPR layout.
__device__ __forceinline__ v16bf load_b_packed(const unsigned* tileBase,
                                               int lane) {
  const uint4* p = (const uint4*)(tileBase + lane * 8);
  uint4 a = p[0], b = p[1];
  v8u u;
  u[0] = a.x; u[1] = a.y; u[2] = a.z; u[3] = a.w;
  u[4] = b.x; u[5] = b.y; u[6] = b.z; u[7] = b.w;
  return __builtin_bit_cast(v16bf, u);
}

// ---------- Pre-pass: x -> bf16 row-major ----------
__global__ __launch_bounds__(256) void pack_x(const float* __restrict__ x,
                                              unsigned* __restrict__ xb) {
  size_t i = (size_t)blockIdx.x * 256 + threadIdx.x;   // one dword = 2 elements
  const size_t total = (size_t)B_TOK * DDIM / 2;
  if (i >= total) return;
  float2 f = ((const float2*)x)[i];
  xb[i] = pack2(f.x, f.y);
}

// ---------- Pre-pass: W[e][K][N] fp32 -> packed bf16 B-fragment tiles ----------
// out[e][N/16][K/32][lane(32)][v(8)] ; value (l,v) = pair at
//   k = kt*32 + (l>=16)*8 + (v>=4)*16 + 2(v&3) , n = nt*16 + (l&15)
__global__ __launch_bounds__(256) void pack_w(const float* __restrict__ W,
                                              unsigned* __restrict__ out,
                                              int K, int N) {
  size_t o = (size_t)blockIdx.x * 256 + threadIdx.x;
  const size_t total = (size_t)NEXP * K * N / 2;
  if (o >= total) return;
  int v = (int)(o & 7);
  int l = (int)((o >> 3) & 31);
  size_t rest = o >> 8;
  int kt = (int)(rest % (K / 32)); rest /= (K / 32);
  int nt = (int)(rest % (N / 16));
  int e  = (int)(rest / (N / 16));
  int k = kt * 32 + ((l >> 4) << 3) + ((v >> 2) << 4) + 2 * (v & 3);
  int n = nt * 16 + (l & 15);
  const float* base = W + ((size_t)e * K + k) * N + n;
  out[o] = pack2(base[0], base[N]);
}

// ---------- Kernel 1: gating (logits + softmax + top2 + scatter) ----------
__global__ __launch_bounds__(256) void moe_gate(
    const float* __restrict__ x, const float* __restrict__ Wg,
    const float* __restrict__ bg, int* __restrict__ counts,
    int* __restrict__ btok, float* __restrict__ bw) {
  const int wave = threadIdx.x >> 5;
  const int lane = threadIdx.x & 31;
  const int t = blockIdx.x * 8 + wave;

  float acc[NEXP];
#pragma unroll
  for (int e = 0; e < NEXP; ++e) acc[e] = 0.0f;

  const float* xr = x + (size_t)t * DDIM;
  for (int d = lane; d < DDIM; d += 32) {
    float xv = xr[d];
    const float4* wr = (const float4*)(Wg + (size_t)d * NEXP);
    float4 w0 = wr[0], w1 = wr[1];
    acc[0] += xv * w0.x; acc[1] += xv * w0.y;
    acc[2] += xv * w0.z; acc[3] += xv * w0.w;
    acc[4] += xv * w1.x; acc[5] += xv * w1.y;
    acc[6] += xv * w1.z; acc[7] += xv * w1.w;
  }
#pragma unroll
  for (int off = 16; off >= 1; off >>= 1) {
#pragma unroll
    for (int e = 0; e < NEXP; ++e) acc[e] += __shfl_xor(acc[e], off, 32);
  }

  if (lane == 0) {
    float mx = -1e30f;
#pragma unroll
    for (int e = 0; e < NEXP; ++e) { acc[e] += bg[e]; mx = fmaxf(mx, acc[e]); }
    float p[NEXP];
#pragma unroll
    for (int e = 0; e < NEXP; ++e) p[e] = __expf(acc[e] - mx);
    int i0 = 0; float b0 = p[0];
#pragma unroll
    for (int e = 1; e < NEXP; ++e) if (p[e] > b0) { b0 = p[e]; i0 = e; }
    int i1 = (i0 == 0) ? 1 : 0; float b1v = p[i1];
#pragma unroll
    for (int e = 0; e < NEXP; ++e)
      if (e != i0 && p[e] > b1v) { b1v = p[e]; i1 = e; }
    float inv = 1.0f / (b0 + b1v);   // softmax normalizer cancels in renorm
    int s0 = atomicAdd(&counts[i0], 1);
    btok[i0 * B_TOK + s0] = t; bw[i0 * B_TOK + s0] = b0 * inv;
    int s1 = atomicAdd(&counts[i1], 1);
    btok[i1 * B_TOK + s1] = t; bw[i1 * B_TOK + s1] = b1v * inv;
  }
}

// ---------- Kernel 2: fused routed expert FFN (bf16 WMMA) ----------
__global__ __launch_bounds__(THREADS, 1) void moe_ffn(
    const unsigned short* __restrict__ xbf, const unsigned* __restrict__ W1p,
    const unsigned* __restrict__ W2p, const float* __restrict__ b1,
    const float* __restrict__ b2, const int* __restrict__ counts,
    const int* __restrict__ btok, const float* __restrict__ bw,
    float* __restrict__ out) {
  const int e    = blockIdx.x / TILES_PER_E;   // consecutive blocks share expert
  const int tile = blockIdx.x % TILES_PER_E;   //  -> expert weights stay hot in L2
  const int cnt  = counts[e];
  const int m0   = tile * TM;
  if (m0 >= cnt) return;                       // block-uniform: EXEC stays all-1s

  __shared__ int            toks[TM];
  __shared__ float          wts[TM];
  __shared__ unsigned short hs[TM][HC];        // bf16 hidden chunk, 16 KB

  if (threadIdx.x < TM) {
    int idx = m0 + threadIdx.x;
    bool valid = idx < cnt;
    toks[threadIdx.x] = btok[e * B_TOK + (valid ? idx : m0)];
    wts[threadIdx.x]  = valid ? bw[e * B_TOK + idx] : 0.0f;   // pad rows: weight 0
  }
  __syncthreads();

  const int wave    = threadIdx.x >> 5;
  const int lane    = threadIdx.x & 31;
  const int lane_hi = lane >> 4;
  const int ln      = lane & 15;
  const int nsub    = wave & 7;        // N-subtile (phase A) / N-slice (phase B)
  const int mp      = (wave >> 3) * 2; // first of this wave's two M-subtiles

  // Per-lane bf16 x rows for the A fragments of the two M-subtiles
  const unsigned short* arow0 = xbf + (size_t)toks[mp * 16 + ln] * DDIM;
  const unsigned short* arow1 = xbf + (size_t)toks[(mp + 1) * 16 + ln] * DDIM;

  // Packed-weight bases for this expert
  const unsigned* W1e = W1p + (size_t)e * (HDIM / 16) * (DDIM / 32) * 256;
  const unsigned* W2e = W2p + (size_t)e * (DDIM / 16) * (HDIM / 32) * 256;

  // y accumulators: rows [mp*16, mp*16+32) x cols [nsub*128, +128)
  v8f yacc[16];
#pragma unroll
  for (int i = 0; i < 16; ++i) yacc[i] = vzero8();

  for (int hc = 0; hc < HDIM; hc += HC) {
    // ---- Phase A: h(64 x 128) = gelu(x_tile @ W1[:, hc:hc+128] + b1) ----
    const int       nt1   = (hc >> 4) + nsub;                 // N-tile in W1p
    const unsigned* bbase = W1e + (size_t)nt1 * (DDIM / 32) * 256;
    v8f c0 = vzero8(), c1 = vzero8();
#pragma unroll 4
    for (int kt = 0; kt < DDIM / 32; ++kt) {
      v16bf bf = load_b_packed(bbase + (size_t)kt * 256, lane);
      v16bf a0 = load_a16(arow0, kt * 32, lane_hi);
      v16bf a1 = load_a16(arow1, kt * 32, lane_hi);
      c0 = wmma_bf16(a0, bf, c0);
      c1 = wmma_bf16(a1, bf, c1);
    }
    const int   ncol  = hc + nsub * 16 + ln;
    const float bias1 = b1[(size_t)e * HDIM + ncol];
    const int   ccol  = nsub * 16 + ln;
#pragma unroll
    for (int r = 0; r < 8; ++r) {
      int mrow = r + 8 * lane_hi;                         // C/D layout (ISA 7.12.2)
      hs[mp * 16 + mrow][ccol]       = f2bf(gelu_exact(c0[r] + bias1));
      hs[(mp + 1) * 16 + mrow][ccol] = f2bf(gelu_exact(c1[r] + bias1));
    }
    __syncthreads();

    // ---- Phase B: y(64 x 1024) += h_chunk @ W2[hc:hc+128, :] ----
    const unsigned short* hrow0 = &hs[mp * 16 + ln][0];
    const unsigned short* hrow1 = &hs[(mp + 1) * 16 + ln][0];
#pragma unroll
    for (int st = 0; st < 8; ++st) {
      const int       nt2   = nsub * 8 + st;              // N-tile in W2p
      const unsigned* b2bas = W2e + ((size_t)nt2 * (HDIM / 32) + (hc >> 5)) * 256;
#pragma unroll
      for (int kq = 0; kq < HC / 32; ++kq) {
        v16bf bf = load_b_packed(b2bas + (size_t)kq * 256, lane);
        v16bf a0 = load_a16(hrow0, kq * 32, lane_hi);
        v16bf a1 = load_a16(hrow1, kq * 32, lane_hi);
        yacc[st]     = wmma_bf16(a0, bf, yacc[st]);
        yacc[8 + st] = wmma_bf16(a1, bf, yacc[8 + st]);
      }
    }
    __syncthreads();   // hs reused next chunk
  }

  // ---- Epilogue: out[token] += w * (y + b2). Exactly two adds/element. ----
#pragma unroll
  for (int ms = 0; ms < 2; ++ms) {
#pragma unroll
    for (int st = 0; st < 8; ++st) {
      int   n     = nsub * 128 + st * 16 + ln;
      float bias2 = b2[(size_t)e * DDIM + n];
      v8f   c     = yacc[ms * 8 + st];
#pragma unroll
      for (int r = 0; r < 8; ++r) {
        int   m = (mp + ms) * 16 + r + 8 * lane_hi;
        float v = (c[r] + bias2) * wts[m];
        atomicAdd(out + (size_t)toks[m] * DDIM + n, v);
      }
    }
  }
}

// ---------- launch ----------
extern "C" void kernel_launch(void* const* d_in, const int* in_sizes, int n_in,
                              void* d_out, int out_size, void* d_ws,
                              size_t ws_size, hipStream_t stream) {
  (void)in_sizes; (void)n_in; (void)ws_size;
  const float* x  = (const float*)d_in[0];
  const float* Wg = (const float*)d_in[1];
  const float* bg = (const float*)d_in[2];
  const float* W1 = (const float*)d_in[3];
  const float* b1 = (const float*)d_in[4];
  const float* W2 = (const float*)d_in[5];
  const float* b2 = (const float*)d_in[6];
  float* out = (float*)d_out;

  // workspace layout (~146 MB):
  //   [0,256)                      counts (E ints)
  //   [256, 256+256K)              btok
  //   [+256K)                      bw
  //   [1M,  17M)                   xbf   (bf16 x, row-major)
  //   [17M, 81M)                   W1p   (packed bf16 B-tiles)
  //   [81M, 145M)                  W2p
  char*  ws     = (char*)d_ws;
  int*   counts = (int*)ws;
  int*   btok   = (int*)(ws + 256);
  float* bw     = (float*)(ws + 256 + (size_t)NEXP * B_TOK * 4);
  unsigned short* xbf = (unsigned short*)(ws + (1ull << 20));
  unsigned* W1p = (unsigned*)(ws + (1ull << 20) + ((size_t)B_TOK * DDIM * 2));
  unsigned* W2p = (unsigned*)((char*)W1p + (size_t)NEXP * DDIM * HDIM * 2);

  hipMemsetAsync(counts, 0, NEXP * sizeof(int), stream);
  hipMemsetAsync(out, 0, (size_t)out_size * sizeof(float), stream);

  {
    size_t total = (size_t)B_TOK * DDIM / 2;
    pack_x<<<(unsigned)((total + 255) / 256), 256, 0, stream>>>(x, (unsigned*)xbf);
  }
  {
    size_t total = (size_t)NEXP * DDIM * HDIM / 2;
    pack_w<<<(unsigned)((total + 255) / 256), 256, 0, stream>>>(W1, W1p, DDIM, HDIM);
    pack_w<<<(unsigned)((total + 255) / 256), 256, 0, stream>>>(W2, W2p, HDIM, DDIM);
  }
  moe_gate<<<B_TOK / 8, 256, 0, stream>>>(x, Wg, bg, counts, btok, bw);
  moe_ffn<<<NEXP * TILES_PER_E, THREADS, 0, stream>>>(xbf, W1p, W2p, b1, b2,
                                                      counts, btok, bw, out);
}
